// PLEv6_6090263626427
// MI455X (gfx1250) — matrix operand order repaired and checked by reading, not compiled, not run on hardware
//
#include <hip/hip_runtime.h>
#include <math.h>

typedef __attribute__((ext_vector_type(16))) _Float16 v16h;
typedef __attribute__((ext_vector_type(8)))  float    v8f;
typedef __attribute__((__vector_size__(16))) int      vint4;

#define BN 8192  // batch

#if __has_builtin(__builtin_amdgcn_global_load_async_to_lds_b128) && \
    __has_builtin(__builtin_amdgcn_s_wait_asynccnt)
#define ASYNC_LDS 1
#endif
#define AS1 __attribute__((address_space(1)))
#define AS3 __attribute__((address_space(3)))

__device__ __forceinline__ float gelu_f(float x) {
    return 0.5f * x * (1.0f + erff(x * 0.70710678118654752f));
}
__device__ __forceinline__ float sigmoid_f(float x) {
    return 1.0f / (1.0f + expf(-x));
}

// ---------------------------------------------------------------------------
// f32 -> f16 convert (grid-strided)
// ---------------------------------------------------------------------------
__global__ void cvt_f16_kernel(const float* __restrict__ s, _Float16* __restrict__ d, int n) {
    int i = blockIdx.x * blockDim.x + threadIdx.x;
    int st = gridDim.x * blockDim.x;
    for (; i < n; i += st) d[i] = (_Float16)s[i];
}

// ---------------------------------------------------------------------------
// Generic WMMA GEMM:  C[z] = act(A[z] @ W[z] + bias[z])
//   A: f16 [M x K], row stride lda, per-z element stride az (M = grid.x*128)
//   W: f16 [K x N] row-major, per-z stride bz
//   Outputs: optional f32 Cf and/or f16 Ch (row stride ldc)
//   act: 0 = none, 1 = exact GELU
// Workgroup tile 128x64, wave tile 32x32 (4 x v_wmma_f32_16x16x32_f16 per
// K-step).  A tile staged via GLOBAL_LOAD_ASYNC_TO_LDS_B128 when available.
// Requires K % 32 == 0; M % 128 == 0; N arbitrary (zero-padded in LDS).
// ---------------------------------------------------------------------------
#define APAD 40   // halves per LDS row (80 B: 16B-aligned & conflict-free)
__global__ __launch_bounds__(256) void gemm_wmma(
    const _Float16* __restrict__ A, int lda, long az,
    const _Float16* __restrict__ Bw, long bz,
    const float* __restrict__ bias, long biasz,
    float* __restrict__ Cf, long cfz,
    _Float16* __restrict__ Ch, long chz,
    int ldc, int N, int K, int act)
{
    __shared__ _Float16 As[128 * APAD];
    __shared__ _Float16 Bs[64 * APAD];

    const int t    = threadIdx.x;
    const int z    = blockIdx.z;
    const int m0   = blockIdx.x * 128;
    const int n0   = blockIdx.y * 64;
    const _Float16* Az = A + (long)z * az;
    const _Float16* Wz = Bw + (long)z * bz;

    const int w     = t >> 5;        // wave 0..7
    const int lane  = t & 31;
    const int halfW = lane >> 4;     // 0/1
    const int l16   = lane & 15;
    const int wm    = w & 3;         // 4 row-blocks of 32
    const int wn    = w >> 2;        // 2 col-blocks of 32

    // loader mapping
    const int ar = t >> 1;           // 0..127 (A row)
    const int ac = (t & 1) * 16;     // 0 or 16 (A col chunk of 16 halves)
    const int bk = t >> 3;           // 0..31  (B k-row)
    const int bn = (t & 7) * 8;      // 0..56  (B col chunk of 8 halves)

    v8f zero = {0.f,0.f,0.f,0.f,0.f,0.f,0.f,0.f};
    v8f acc[2][2];
    acc[0][0] = zero; acc[0][1] = zero; acc[1][0] = zero; acc[1][1] = zero;

    for (int kk = 0; kk < K; kk += 32) {
        // stage A tile [128 x 32] (async global->LDS when available)
        {
            const _Float16* src = Az + (long)(m0 + ar) * lda + kk + ac;
            _Float16* dst = &As[ar * APAD + ac];
#ifdef ASYNC_LDS
            __builtin_amdgcn_global_load_async_to_lds_b128(
                (AS1 vint4*)(void*)src, (AS3 vint4*)(void*)dst, 0, 0);
            __builtin_amdgcn_global_load_async_to_lds_b128(
                (AS1 vint4*)(void*)(src + 8), (AS3 vint4*)(void*)(dst + 8), 0, 0);
#else
            uint4 v0 = *reinterpret_cast<const uint4*>(src);
            uint4 v1 = *reinterpret_cast<const uint4*>(src + 8);
            *reinterpret_cast<uint4*>(dst) = v0;
            *reinterpret_cast<uint4*>(dst + 8) = v1;
#endif
        }
        // stage B tile transposed -> Bs[n][k]
        {
            _Float16 tmp[8];
            if (n0 + bn < N) {
                uint4 v = *reinterpret_cast<const uint4*>(Wz + (long)(kk + bk) * N + n0 + bn);
                *reinterpret_cast<uint4*>(tmp) = v;
            } else {
                #pragma unroll
                for (int j = 0; j < 8; ++j) tmp[j] = (_Float16)0.f;
            }
            #pragma unroll
            for (int j = 0; j < 8; ++j) Bs[(bn + j) * APAD + bk] = tmp[j];
        }
        if (kk + 32 < K) {
            __builtin_prefetch(Az + (long)(m0 + ar) * lda + (kk + 32) + ac, 0, 3);
            if (n0 + bn < N)
                __builtin_prefetch(Wz + (long)(kk + 32 + bk) * N + n0 + bn, 0, 3);
        }
#ifdef ASYNC_LDS
        __builtin_amdgcn_s_wait_asynccnt(0);
#endif
        __syncthreads();

        // fragments per documented gfx1250 16-bit layouts
        union Frag { v16h v; uint u[8]; };
        Frag af[2], bf[2];
        #pragma unroll
        for (int rf = 0; rf < 2; ++rf) {
            const int arow = (wm * 32 + rf * 16 + l16) * APAD;
            #pragma unroll
            for (int v2 = 0; v2 < 8; ++v2) {
                int kb = ((v2 & 4) ? 16 : 0) + halfW * 8 + (v2 & 3) * 2;
                af[rf].u[v2] = *reinterpret_cast<const uint*>(&As[arow + kb]);
            }
        }
        const int klo = halfW * 8;
        #pragma unroll
        for (int f = 0; f < 2; ++f) {
            const int brow = (wn * 32 + f * 16 + l16) * APAD + klo;
            #pragma unroll
            for (int v2 = 0; v2 < 8; ++v2)
                bf[f].u[v2] = *reinterpret_cast<const uint*>(&Bs[brow + ((v2 & 4) ? 16 : 0) + (v2 & 3) * 2]);
        }
        #pragma unroll
        for (int rf = 0; rf < 2; ++rf)
            #pragma unroll
            for (int f = 0; f < 2; ++f)
                acc[rf][f] = __builtin_amdgcn_wmma_f32_16x16x32_f16(
                    false, af[rf].v, false, bf[f].v, (short)0, acc[rf][f], false, false);
        __syncthreads();
    }

    // epilogue: bias + activation, write f32 and/or f16
    float* cf = Cf ? Cf + (long)z * cfz : nullptr;
    _Float16* ch = Ch ? Ch + (long)z * chz : nullptr;
    const float* bp = bias ? bias + (long)z * biasz : nullptr;
    #pragma unroll
    for (int rf = 0; rf < 2; ++rf) {
        #pragma unroll
        for (int f = 0; f < 2; ++f) {
            int col = n0 + wn * 32 + f * 16 + l16;
            if (col < N) {
                float bv = bp ? bp[col] : 0.f;
                #pragma unroll
                for (int v2 = 0; v2 < 8; ++v2) {
                    float x = acc[rf][f][v2] + bv;
                    if (act == 1) x = gelu_f(x);
                    long row = m0 + wm * 32 + rf * 16 + v2 + halfW * 8;
                    if (cf) cf[row * (long)ldc + col] = x;
                    if (ch) ch[row * (long)ldc + col] = (_Float16)x;
                }
            }
        }
    }
}

// ---------------------------------------------------------------------------
// LayerNorm over last dim N (<=512), optional residual, optional f32/f16 out.
// grid = (rows, Z), block = 256
// ---------------------------------------------------------------------------
__global__ __launch_bounds__(256) void ln_kernel(
    const float* __restrict__ X, long xz,
    const float* __restrict__ RES, long rz,
    const float* __restrict__ G, const float* __restrict__ Bt, long gz,
    float* __restrict__ Yf, long yfz, int ldyf,
    _Float16* __restrict__ Yh, long yhz, int ldyh,
    int N)
{
    __shared__ float red[256];
    const int r = blockIdx.x, z = blockIdx.y, t = threadIdx.x;
    const float* xp = X + (long)z * xz + (long)r * N;
    const float* rp = RES ? RES + (long)z * rz + (long)r * N : nullptr;
    float xv[2] = {0.f, 0.f};
    const int cnt = (N + 255) >> 8;
    float s = 0.f;
    for (int i = 0; i < cnt; ++i) {
        int c = t + i * 256;
        if (c < N) { xv[i] = xp[c] + (rp ? rp[c] : 0.f); s += xv[i]; }
    }
    red[t] = s; __syncthreads();
    for (int k = 128; k; k >>= 1) { if (t < k) red[t] += red[t + k]; __syncthreads(); }
    const float mean = red[0] / (float)N;
    __syncthreads();
    float s2 = 0.f;
    for (int i = 0; i < cnt; ++i) {
        int c = t + i * 256;
        if (c < N) { float d = xv[i] - mean; s2 += d * d; }
    }
    red[t] = s2; __syncthreads();
    for (int k = 128; k; k >>= 1) { if (t < k) red[t] += red[t + k]; __syncthreads(); }
    const float inv = rsqrtf(red[0] / (float)N + 1e-5f);
    for (int i = 0; i < cnt; ++i) {
        int c = t + i * 256;
        if (c < N) {
            float y = (xv[i] - mean) * inv * G[(long)z * gz + c] + Bt[(long)z * gz + c];
            if (Yf) Yf[(long)z * yfz + (long)r * ldyf + c] = y;
            if (Yh) Yh[(long)z * yhz + (long)r * ldyh + c] = (_Float16)y;
        }
    }
}

// ---------------------------------------------------------------------------
// temporal encoder: [B,40] -> 64 (gelu) -> 32, LN.  block = 32
// ---------------------------------------------------------------------------
__global__ void temporal_kernel(
    const float* __restrict__ T,
    const float* __restrict__ W1, const float* __restrict__ b1,
    const float* __restrict__ W2, const float* __restrict__ b2,
    const float* __restrict__ g, const float* __restrict__ be,
    float* __restrict__ out)
{
    __shared__ float tin[40];
    __shared__ float h1[64];
    const int r = blockIdx.x, j = threadIdx.x;
    tin[j] = T[(long)r * 40 + j];
    if (j < 8) tin[32 + j] = T[(long)r * 40 + 32 + j];
    __syncthreads();
    for (int rep = 0; rep < 2; ++rep) {
        int o = j + rep * 32;
        float s = b1[o];
        for (int k = 0; k < 40; ++k) s += tin[k] * W1[k * 64 + o];
        h1[o] = gelu_f(s);
    }
    __syncthreads();
    float s2 = b2[j];
    for (int k = 0; k < 64; ++k) s2 += h1[k] * W2[k * 32 + j];
    float m = s2;
    for (int off = 16; off; off >>= 1) m += __shfl_xor(m, off);
    m *= (1.f / 32.f);
    float d = s2 - m;
    float v = d * d;
    for (int off = 16; off; off >>= 1) v += __shfl_xor(v, off);
    v *= (1.f / 32.f);
    out[(long)r * 32 + j] = d * rsqrtf(v + 1e-5f) * g[j] + be[j];
}

// ---------------------------------------------------------------------------
// context encoder: gelu(cat(account,coin,regime,temporal) @ W + b) -> f16 slices
// block = 64
// ---------------------------------------------------------------------------
__global__ void ctx_kernel(
    const float* __restrict__ acct, const int* __restrict__ coin_id,
    const int* __restrict__ regime_id,
    const float* __restrict__ coinE, const float* __restrict__ regE,
    const float* __restrict__ tenc,
    const float* __restrict__ W, const float* __restrict__ bb,
    _Float16* __restrict__ allcat16, _Float16* __restrict__ fus16)
{
    __shared__ float xin[84];
    const int r = blockIdx.x, j = threadIdx.x;
    if (j < 4)  xin[j]      = acct[(long)r * 4 + j];
    if (j < 32) xin[4 + j]  = coinE[(long)coin_id[r] * 32 + j];
    if (j < 16) xin[36 + j] = regE[(long)regime_id[r] * 16 + j];
    if (j < 32) xin[52 + j] = tenc[(long)r * 32 + j];
    __syncthreads();
    float s = bb[j];
    for (int k = 0; k < 84; ++k) s += xin[k] * W[k * 64 + j];
    s = gelu_f(s);
    allcat16[(long)r * 1088 + 1024 + j] = (_Float16)s;
    fus16[(long)r * 320 + 256 + j] = (_Float16)s;
}

// ---------------------------------------------------------------------------
// gate: scores, softmax, weighted combine.  block = 256
// ---------------------------------------------------------------------------
__global__ __launch_bounds__(256) void gate_kernel(
    const float* __restrict__ q, const float* __restrict__ kout,
    const float* __restrict__ allcat, float* __restrict__ gated,
    _Float16* __restrict__ gated16, float* __restrict__ gw_out)
{
    __shared__ float red[256];
    __shared__ float wsm[4];
    const int r = blockIdx.x, n = threadIdx.x;
    const float qv = q[(long)r * 256 + n];
    float sc[4];
    for (int p = 0; p < 4; ++p) {
        red[n] = qv * kout[((long)p * BN + r) * 256 + n];
        __syncthreads();
        for (int k = 128; k; k >>= 1) { if (n < k) red[n] += red[n + k]; __syncthreads(); }
        sc[p] = red[0];
        __syncthreads();
    }
    if (n == 0) {
        float mx = -1e30f;
        for (int p = 0; p < 4; ++p) { sc[p] *= (1.f / 16.f); if (sc[p] > mx) mx = sc[p]; }
        float ssum = 0.f, e[4];
        for (int p = 0; p < 4; ++p) { e[p] = expf(sc[p] - mx); ssum += e[p]; }
        for (int p = 0; p < 4; ++p) { wsm[p] = e[p] / ssum; gw_out[(long)r * 4 + p] = wsm[p]; }
    }
    __syncthreads();
    float gv = 0.f;
    for (int p = 0; p < 4; ++p) gv += wsm[p] * allcat[(long)r * 1088 + p * 256 + n];
    gated[(long)r * 256 + n] = gv;
    gated16[(long)r * 256 + n] = (_Float16)gv;
}

// ---------------------------------------------------------------------------
// router: gelu(cat(gated,regime)@W1+b1)@W2+b2, top-2 + softmax.  block = 32
// ---------------------------------------------------------------------------
__global__ void router_kernel(
    const float* __restrict__ gated, const int* __restrict__ regime_id,
    const float* __restrict__ regE,
    const float* __restrict__ W1, const float* __restrict__ b1,
    const float* __restrict__ W2, const float* __restrict__ b2,
    float* __restrict__ out_moew, float* __restrict__ out_topi,
    int* __restrict__ topi)
{
    __shared__ float xin[272];
    __shared__ float u[16];
    __shared__ float lg[8];
    const int r = blockIdx.x, j = threadIdx.x;
    for (int k = j; k < 256; k += 32) xin[k] = gated[(long)r * 256 + k];
    if (j < 16) xin[256 + j] = regE[(long)regime_id[r] * 16 + j];
    __syncthreads();
    if (j < 16) {
        float s = b1[j];
        for (int k = 0; k < 272; ++k) s += xin[k] * W1[k * 16 + j];
        u[j] = gelu_f(s);
    }
    __syncthreads();
    if (j < 8) {
        float s = b2[j];
        for (int k = 0; k < 16; ++k) s += u[k] * W2[k * 8 + j];
        lg[j] = s;
    }
    __syncthreads();
    if (j == 0) {
        int i0 = 0; float v0 = lg[0];
        for (int k = 1; k < 8; ++k) if (lg[k] > v0) { v0 = lg[k]; i0 = k; }
        int i1 = -1; float v1 = -1e30f;
        for (int k = 0; k < 8; ++k) if (k != i0 && lg[k] > v1) { v1 = lg[k]; i1 = k; }
        float e0 = expf(v0 - v0), e1 = expf(v1 - v0);
        float ss = e0 + e1;
        out_moew[(long)r * 2 + 0] = e0 / ss;
        out_moew[(long)r * 2 + 1] = e1 / ss;
        out_topi[(long)r * 2 + 0] = (float)i0;
        out_topi[(long)r * 2 + 1] = (float)i1;
        topi[r * 2 + 0] = i0;
        topi[r * 2 + 1] = i1;
    }
}

// ---------------------------------------------------------------------------
// MoE top-2 combine -> fusion input cols 0..255 (f16).  block = 256
// ---------------------------------------------------------------------------
__global__ __launch_bounds__(256) void moe_combine_kernel(
    const float* __restrict__ eo, const int* __restrict__ topi,
    const float* __restrict__ moew, _Float16* __restrict__ fus16)
{
    const int r = blockIdx.x, n = threadIdx.x;
    const int i0 = topi[r * 2], i1 = topi[r * 2 + 1];
    const float w0 = moew[(long)r * 2], w1 = moew[(long)r * 2 + 1];
    float v = w0 * eo[((long)i0 * BN + r) * 256 + n] +
              w1 * eo[((long)i1 * BN + r) * 256 + n];
    fus16[(long)r * 320 + n] = (_Float16)v;
}

// ---------------------------------------------------------------------------
// head label tails: logits + sigmoid.  block = 32 (4 heads x 8 labels)
// ---------------------------------------------------------------------------
__global__ void lab2_kernel(
    const float* __restrict__ hlab, const float* __restrict__ w,
    const float* __restrict__ b, float* __restrict__ o_logits,
    float* __restrict__ o_probs)
{
    const int r = blockIdx.x, t = threadIdx.x;
    const int g = t >> 3, o = t & 7;
    const float* h = hlab + ((long)g * BN + r) * 256;
    const float* wg = w + (long)g * 256 * 8;
    float s = b[g * 8 + o];
    for (int k = 0; k < 256; ++k) s += h[k] * wg[k * 8 + o];
    o_logits[(long)r * 32 + t] = s;
    o_probs[(long)r * 32 + t] = sigmoid_f(s);
}

// ---------------------------------------------------------------------------
// mae/mfe tails. block = 64 (z 0..3 = mae heads, 4..7 = mfe heads)
// ---------------------------------------------------------------------------
__global__ void maefe2_kernel(
    const float* __restrict__ hm, const float* __restrict__ w,
    const float* __restrict__ b, float* __restrict__ o_mae,
    float* __restrict__ o_mfe)
{
    const int r = blockIdx.x, t = threadIdx.x;
    const int z = t >> 3, o = t & 7;
    const float* h = hm + ((long)z * BN + r) * 32;
    const float* wz = w + (long)z * 32 * 8;
    float s = b[z * 8 + o];
    for (int k = 0; k < 32; ++k) s += h[k] * wz[k * 8 + o];
    if (z < 4) o_mae[(long)r * 32 + z * 8 + o] = s;
    else       o_mfe[(long)r * 32 + (z - 4) * 8 + o] = s;
}

// ---------------------------------------------------------------------------
// confidence / leverage tails. block = 64 (wave 0 = conf, wave 1 = lev)
// ---------------------------------------------------------------------------
__global__ void conflev_kernel(
    const float* __restrict__ hc,
    const float* __restrict__ wC, const float* __restrict__ bC,
    const float* __restrict__ wL, const float* __restrict__ bL,
    float* __restrict__ outC, float* __restrict__ outL)
{
    const int r = blockIdx.x, t = threadIdx.x;
    const int z = t >> 5, k = t & 31;
    const float* h = hc + ((long)z * BN + r) * 32;
    const float* w = z ? wL : wC;
    float v = h[k] * w[k];
    for (int off = 16; off; off >>= 1) v += __shfl_xor(v, off);
    if (k == 0) {
        float s = v + (z ? bL[0] : bC[0]);
        s = sigmoid_f(s);
        if (z) outL[r] = s; else outC[r] = s;
    }
}

// ---------------------------------------------------------------------------
// host orchestration
// ---------------------------------------------------------------------------
extern "C" void kernel_launch(void* const* d_in, const int* in_sizes, int n_in,
                              void* d_out, int out_size, void* d_ws, size_t ws_size,
                              hipStream_t stream) {
    (void)in_sizes; (void)n_in; (void)out_size; (void)ws_size;
    const long B_ = BN;
    float* out = (float*)d_out;

    // input index map (setup_inputs dict order, params flattened depth-first)
    const float* features = (const float*)d_in[0];
    const int*   coin_id  = (const int*)d_in[1];
    const int*   regime_id= (const int*)d_in[2];
    const float* account  = (const float*)d_in[3];
    const float* temporal = (const float*)d_in[4];
    auto F = [&](int i) -> const float* { return (const float*)d_in[i]; };
    const int P_FEAT = 13, P_GK = 57, P_HEAD = 85;

    // output regions (tuple order, flat)
    float* o_logits = out;
    float* o_probs  = out + B_ * 32;
    float* o_mae    = out + 2 * B_ * 32;
    float* o_mfe    = out + 3 * B_ * 32;
    float* o_conf   = out + 4 * B_ * 32;
    float* o_lev    = o_conf + B_;
    float* o_gatew  = o_lev + B_;
    float* o_moew   = o_gatew + B_ * 4;
    float* o_topi   = o_moew + B_ * 2;

    // ---- workspace bump allocator ----
    size_t off = 0;
    auto alloc = [&](size_t bytes) -> char* {
        off = (off + 255) & ~(size_t)255;
        char* p = (char*)d_ws + off;
        off += bytes;
        return p;
    };
    // packed f16 weights + f32 params
    _Float16* wFeatL1 = (_Float16*)alloc((size_t)4*32*512*2);
    _Float16* wFeatL2 = (_Float16*)alloc((size_t)4*512*512*2);
    _Float16* wFeatL3 = (_Float16*)alloc((size_t)4*512*256*2);
    _Float16* wFeatR  = (_Float16*)alloc((size_t)4*32*256*2);
    float* bFeatL1 = (float*)alloc(4*512*4);
    float* bFeatL2 = (float*)alloc(4*512*4);
    float* bFeatL3 = (float*)alloc(4*256*4);
    float* bFeatR  = (float*)alloc(4*256*4);
    float* gFeat   = (float*)alloc(4*256*4);
    float* betaFeat= (float*)alloc(4*256*4);
    _Float16* wGQ  = (_Float16*)alloc((size_t)1088*256*2);
    _Float16* wGK  = (_Float16*)alloc((size_t)4*256*256*2);
    float* bGK     = (float*)alloc(4*256*4);
    _Float16* wM1  = (_Float16*)alloc((size_t)8*256*512*2);
    _Float16* wM2  = (_Float16*)alloc((size_t)8*512*512*2);
    _Float16* wM3  = (_Float16*)alloc((size_t)8*512*256*2);
    _Float16* wF1  = (_Float16*)alloc((size_t)320*512*2);
    _Float16* wF2  = (_Float16*)alloc((size_t)512*512*2);
    _Float16* wLab1= (_Float16*)alloc((size_t)4*512*256*2);
    float* bLab1   = (float*)alloc(4*256*4);
    float* wLab2   = (float*)alloc(4*256*8*4);
    float* bLab2   = (float*)alloc(4*8*4);
    _Float16* wMF1 = (_Float16*)alloc((size_t)8*512*32*2);
    float* bMF1    = (float*)alloc(8*32*4);
    float* wMF2    = (float*)alloc(8*32*8*4);
    float* bMF2    = (float*)alloc(8*8*4);
    _Float16* wCL1 = (_Float16*)alloc((size_t)2*512*32*2);
    float* bCL1    = (float*)alloc(2*32*4);
    // activations (with lifetime-based reuse)
    _Float16* feat16 = (_Float16*)alloc((size_t)B_*128*2);
    char* R1 = alloc((size_t)8*B_*512*2);                // feat h1/h2 -> moe h1
    _Float16* h1_16 = (_Float16*)R1;
    _Float16* h2_16 = (_Float16*)R1 + (size_t)4*B_*512;
    _Float16* mh1_16 = (_Float16*)R1;
    char* R2 = alloc((size_t)8*B_*512*2);                // feat y/yr -> moe h2
    float* y_f  = (float*)R2;
    float* yr_f = (float*)(R2 + (size_t)4*B_*256*4);
    _Float16* mh2_16 = (_Float16*)R2;
    char* R3 = alloc((size_t)B_*1088*4 + (size_t)B_*256*4 + (size_t)4*B_*256*4); // allcat/q/kout -> moe y
    float* allcat_f = (float*)R3;
    float* q_f    = allcat_f + (size_t)B_*1088;
    float* kout_f = q_f + (size_t)B_*256;
    float* my_f   = (float*)R3;
    char* R4 = alloc((size_t)8*B_*256*4);                // allcat16 -> eo -> hlab/hm/hc
    _Float16* allcat16 = (_Float16*)R4;
    float* eo_f   = (float*)R4;
    float* hlab_f = (float*)R4;
    float* hm_f   = (float*)(R4 + (size_t)4*B_*256*4);
    float* hc_f   = (float*)(R4 + (size_t)4*B_*256*4 + (size_t)8*B_*32*4);
    float* gated_f = (float*)alloc((size_t)B_*256*4);
    _Float16* gated16 = (_Float16*)alloc((size_t)B_*256*2);
    float* tenc_f = (float*)alloc((size_t)B_*32*4);
    _Float16* fus_in16 = (_Float16*)alloc((size_t)B_*320*2);
    float* ff_f = (float*)alloc((size_t)B_*512*4);       // fusion h1 then h2
    _Float16* ffl1_16 = (_Float16*)alloc((size_t)B_*512*2);
    _Float16* fused16 = (_Float16*)alloc((size_t)B_*512*2);
    int* topi = (int*)alloc((size_t)B_*2*4);

    auto cvt = [&](const float* src, _Float16* dst, int n) {
        int blocks = (n + 255) / 256; if (blocks > 4096) blocks = 4096;
        cvt_f16_kernel<<<blocks, 256, 0, stream>>>(src, dst, n);
    };
    auto d2d = [&](void* dst, const void* src, size_t bytes) {
        (void)hipMemcpyAsync(dst, src, bytes, hipMemcpyDeviceToDevice, stream);
    };
    auto gemm = [&](const _Float16* A, int lda, long az,
                    const _Float16* W, long bz,
                    const float* bias, long biasz,
                    float* Cf, long cfz, _Float16* Ch, long chz,
                    int ldc, int N, int K, int Z, int act) {
        dim3 grid((unsigned)(B_ / 128), (unsigned)((N + 63) / 64), (unsigned)Z);
        gemm_wmma<<<grid, 256, 0, stream>>>(A, lda, az, W, bz, bias, biasz,
                                            Cf, cfz, Ch, chz, ldc, N, K, act);
    };

    // ---- weight conversion & packing ----
    for (int p = 0; p < 4; ++p) {
        int b0 = P_FEAT + p * 10;
        cvt(F(b0+0), wFeatL1 + (size_t)p*32*512, 32*512);
        cvt(F(b0+2), wFeatL2 + (size_t)p*512*512, 512*512);
        cvt(F(b0+4), wFeatL3 + (size_t)p*512*256, 512*256);
        cvt(F(b0+8), wFeatR  + (size_t)p*32*256, 32*256);
        d2d(bFeatL1 + p*512, F(b0+1), 512*4);
        d2d(bFeatL2 + p*512, F(b0+3), 512*4);
        d2d(bFeatL3 + p*256, F(b0+5), 256*4);
        d2d(gFeat   + p*256, F(b0+6), 256*4);
        d2d(betaFeat+ p*256, F(b0+7), 256*4);
        d2d(bFeatR  + p*256, F(b0+9), 256*4);
    }
    cvt(F(55), wGQ, 1088*256);
    for (int p = 0; p < 4; ++p) {
        cvt(F(P_GK + p*2), wGK + (size_t)p*256*256, 256*256);
        d2d(bGK + p*256, F(P_GK + 1 + p*2), 256*4);
    }
    cvt(F(69), wM1, 8*256*512);
    cvt(F(71), wM2, 8*512*512);
    cvt(F(73), wM3, 8*512*256);
    cvt(F(77), wF1, 320*512);
    cvt(F(81), wF2, 512*512);
    for (int g = 0; g < 4; ++g) {
        int hb = P_HEAD + g * 12;
        cvt(F(hb+0), wLab1 + (size_t)g*512*256, 512*256);
        d2d(bLab1 + g*256, F(hb+1), 256*4);
        d2d(wLab2 + (size_t)g*256*8, F(hb+2), 256*8*4);
        d2d(bLab2 + g*8, F(hb+3), 8*4);
        cvt(F(hb+4), wMF1 + (size_t)g*512*32, 512*32);
        d2d(bMF1 + g*32, F(hb+5), 32*4);
        d2d(wMF2 + (size_t)g*32*8, F(hb+6), 32*8*4);
        d2d(bMF2 + g*8, F(hb+7), 8*4);
        cvt(F(hb+8), wMF1 + (size_t)(4+g)*512*32, 512*32);
        d2d(bMF1 + (4+g)*32, F(hb+9), 32*4);
        d2d(wMF2 + (size_t)(4+g)*32*8, F(hb+10), 32*8*4);
        d2d(bMF2 + (4+g)*8, F(hb+11), 8*4);
    }
    cvt(F(133), wCL1, 512*32);           d2d(bCL1, F(134), 32*4);
    cvt(F(137), wCL1 + 512*32, 512*32);  d2d(bCL1 + 32, F(138), 32*4);
    cvt(features, feat16, (int)(B_ * 128));

    // ---- small encoders ----
    temporal_kernel<<<(unsigned)B_, 32, 0, stream>>>(
        temporal, F(7), F(8), F(9), F(10), F(11), F(12), tenc_f);
    ctx_kernel<<<(unsigned)B_, 64, 0, stream>>>(
        account, coin_id, regime_id, F(5), F(6), tenc_f, F(53), F(54),
        allcat16, fus_in16);

    // ---- feature experts (z = 4 partitions) ----
    gemm(feat16, 128, 32, wFeatL1, 32*512, bFeatL1, 512,
         nullptr, 0, h1_16, (long)B_*512, 512, 512, 32, 4, 1);
    gemm(h1_16, 512, (long)B_*512, wFeatL2, 512*512, bFeatL2, 512,
         nullptr, 0, h2_16, (long)B_*512, 512, 512, 512, 4, 1);
    gemm(h2_16, 512, (long)B_*512, wFeatL3, 512*256, bFeatL3, 256,
         y_f, (long)B_*256, nullptr, 0, 256, 256, 512, 4, 0);
    gemm(feat16, 128, 32, wFeatR, 32*256, bFeatR, 256,
         yr_f, (long)B_*256, nullptr, 0, 256, 256, 32, 4, 0);
    { dim3 grid((unsigned)B_, 4);
      ln_kernel<<<grid, 256, 0, stream>>>(y_f, (long)B_*256, yr_f, (long)B_*256,
          gFeat, betaFeat, 256, allcat_f, 256, 1088, allcat16, 256, 1088, 256); }

    // ---- gating ----
    gemm(allcat16, 1088, 0, wGQ, 0, F(56), 0,
         q_f, 0, nullptr, 0, 256, 256, 1088, 1, 0);
    gemm(allcat16, 1088, 256, wGK, 256*256, bGK, 256,
         kout_f, (long)B_*256, nullptr, 0, 256, 256, 256, 4, 0);
    gate_kernel<<<(unsigned)B_, 256, 0, stream>>>(
        q_f, kout_f, allcat_f, gated_f, gated16, o_gatew);
    router_kernel<<<(unsigned)B_, 32, 0, stream>>>(
        gated_f, regime_id, F(6), F(65), F(66), F(67), F(68),
        o_moew, o_topi, topi);

    // ---- MoE (z = 8 experts, dense) ----
    gemm(gated16, 256, 0, wM1, 256*512, F(70), 512,
         nullptr, 0, mh1_16, (long)B_*512, 512, 512, 256, 8, 1);
    gemm(mh1_16, 512, (long)B_*512, wM2, 512*512, F(72), 512,
         nullptr, 0, mh2_16, (long)B_*512, 512, 512, 512, 8, 1);
    gemm(mh2_16, 512, (long)B_*512, wM3, 512*256, F(74), 256,
         my_f, (long)B_*256, nullptr, 0, 256, 256, 512, 8, 0);
    { dim3 grid((unsigned)B_, 8);
      ln_kernel<<<grid, 256, 0, stream>>>(my_f, (long)B_*256, gated_f, 0,
          F(75), F(76), 256, eo_f, (long)B_*256, 256, nullptr, 0, 0, 256); }
    moe_combine_kernel<<<(unsigned)B_, 256, 0, stream>>>(eo_f, topi, o_moew, fus_in16);

    // ---- fusion ----
    gemm(fus_in16, 320, 0, wF1, 0, F(78), 0,
         ff_f, 0, nullptr, 0, 512, 512, 320, 1, 1);
    { dim3 grid((unsigned)B_, 1);
      ln_kernel<<<grid, 256, 0, stream>>>(ff_f, 0, nullptr, 0, F(79), F(80), 0,
          nullptr, 0, 0, ffl1_16, 0, 512, 512); }
    gemm(ffl1_16, 512, 0, wF2, 0, F(82), 0,
         ff_f, 0, nullptr, 0, 512, 512, 512, 1, 1);
    { dim3 grid((unsigned)B_, 1);
      ln_kernel<<<grid, 256, 0, stream>>>(ff_f, 0, nullptr, 0, F(83), F(84), 0,
          nullptr, 0, 0, fused16, 0, 512, 512); }

    // ---- heads ----
    gemm(fused16, 512, 0, wLab1, 512*256, bLab1, 256,
         hlab_f, (long)B_*256, nullptr, 0, 256, 256, 512, 4, 1);
    lab2_kernel<<<(unsigned)B_, 32, 0, stream>>>(hlab_f, wLab2, bLab2, o_logits, o_probs);
    gemm(fused16, 512, 0, wMF1, 512*32, bMF1, 32,
         hm_f, (long)B_*32, nullptr, 0, 32, 32, 512, 8, 1);
    maefe2_kernel<<<(unsigned)B_, 64, 0, stream>>>(hm_f, wMF2, bMF2, o_mae, o_mfe);
    gemm(fused16, 512, 0, wCL1, 512*32, bCL1, 32,
         hc_f, (long)B_*32, nullptr, 0, 32, 32, 512, 2, 1);
    conflev_kernel<<<(unsigned)B_, 64, 0, stream>>>(hc_f, F(135), F(136), F(139), F(140),
                                                    o_conf, o_lev);
}
